// FlexibleBitterLLM_88295937671425
// MI455X (gfx1250) — matrix operand order repaired and checked by reading, compiled, not run on hardware
//
#include <hip/hip_runtime.h>
#include <math.h>

// ---------------- model constants ----------------
#define B_    4
#define S_    1024
#define V_    256
#define E_    768
#define H_    12
#define D_    64
#define SW_   64
#define BS_   (B_*S_)
#define IMID_ 3072
#define NEGINF (-1e30f)

// ---------------- WMMA types ----------------
typedef __bf16 bf16;
typedef __attribute__((ext_vector_type(16))) __bf16 bf16x16;
typedef __attribute__((ext_vector_type(8)))  float   floatx8;
typedef __attribute__((ext_vector_type(4)))  unsigned int v4u;
typedef __attribute__((ext_vector_type(8)))  int v8i;
typedef __attribute__((ext_vector_type(4)))  int v4i;

#if defined(__has_builtin)
#if __has_builtin(__builtin_amdgcn_tensor_load_to_lds)
#define HAS_TDM 1
#endif
#endif

__device__ __forceinline__ floatx8 zero8() {
  floatx8 z;
#pragma unroll
  for (int i = 0; i < 8; i++) z[i] = 0.f;
  return z;
}

__device__ __forceinline__ floatx8 wmma_bf16(bf16x16 a, bf16x16 b, floatx8 c) {
  return __builtin_amdgcn_wmma_f32_16x16x32_bf16(false, a, false, b, (short)0, c, false, false);
}

__device__ __forceinline__ bf16 f2bf(float f) {
  union { float f; unsigned int u; } in; in.f = f;
  unsigned int u = in.u;
  unsigned int r = u + 0x7fffu + ((u >> 16) & 1u);   // RNE
  union { unsigned short s; bf16 h; } out;
  out.s = (unsigned short)(r >> 16);
  return out.h;
}

__device__ __forceinline__ float gelu_tanh(float x) {
  float x3 = x * x * x;
  return 0.5f * x * (1.f + tanhf(0.7978845608028654f * (x + 0.044715f * x3)));
}

// A fragment (16x32 bf16): lanes 0-15 M=lane (K 0..7,16..23), lanes 16-31 (K 8..15,24..31).
__device__ __forceinline__ bf16x16 frag_a(const bf16* lds, int lda, int m0, int k0, int lane) {
  const int m  = m0 + (lane & 15);
  const int kh = (lane >> 4) << 3;
  const unsigned short* row = (const unsigned short*)(lds + m * lda + k0);
  union { unsigned int u[8]; bf16x16 v; } f;
#pragma unroll
  for (int p = 0; p < 8; p++) {
    const int k = ((p < 4) ? 0 : 16) + ((p & 3) << 1) + kh;
    f.u[p] = *(const unsigned int*)(row + k);
  }
  return f.v;
}

// B fragment (32x16 bf16) from LDS stored [N][K] (K contiguous).
__device__ __forceinline__ bf16x16 frag_b(const bf16* lds, int ldb, int n0, int k0, int lane) {
  const int n  = n0 + (lane & 15);
  const int kh = (lane >> 4) << 3;
  const unsigned short* col = (const unsigned short*)(lds + n * ldb + k0 + kh);
  union { unsigned int u[8]; bf16x16 v; } f;
#pragma unroll
  for (int p = 0; p < 8; p++) f.u[p] = *(const unsigned int*)(col + ((p < 4) ? 0 : 16) + ((p & 3) << 1));
  return f.v;
}

// ---------------- CDNA5 data movers ----------------
__device__ __forceinline__ void async_b128(unsigned lds_addr, const void* g) {
  asm volatile("global_load_async_to_lds_b128 %0, %1, off"
               :: "v"(lds_addr), "v"(g) : "memory");
}
__device__ __forceinline__ void wait_async0() {
  asm volatile("s_wait_asynccnt 0x0" ::: "memory");
}

#define BM  128
#define BN  128
#define BK  32
#define LDP 40   // 80B LDS pitch: 64B tile row + 16B pad (16B aligned, 20-bank stride)

#ifdef HAS_TDM
// TDM: move one 128x32 bf16 tile (row stride `stride` elems) to LDS at lds_addr,
// hardware-padding each 64B row with 16B (-> LDP pitch). Tracked by TENSORcnt.
__device__ __forceinline__ void tdm_load_tile(const bf16* g, unsigned lds_addr, int stride) {
  unsigned long long ga = (unsigned long long)(size_t)g;
  v4u g0;
  g0[0] = 1u;                                               // count=1, user mode
  g0[1] = lds_addr;                                         // lds_addr
  g0[2] = (unsigned)(ga & 0xffffffffu);                     // global_addr[31:0]
  g0[3] = (unsigned)((ga >> 32) & 0x01ffffffu) | (2u << 30); // addr[56:32], type=2
  v8i g1;
  g1[0] = (1 << 16)      // data_size = 2B
        | (1 << 20)      // pad_enable
        | (3 << 22)      // pad_interval: 16 DWORDs (one 64B row)
        | (3 << 25);     // pad_amount: 4 DWORDs (16B)
  g1[1] = (32 & 0xffff) << 16;            // tensor_dim0 lo (=tile_dim0)
  g1[2] = (BM & 0xffff) << 16;            // tensor_dim0 hi=0 | tensor_dim1 lo (=rows)
  g1[3] = (32 << 16);                     // tensor_dim1 hi=0 | tile_dim0 = 32
  g1[4] = BM;                             // tile_dim1 = 128, tile_dim2 = 0
  g1[5] = stride;                         // tensor_dim0_stride[31:0]
  g1[6] = 0;                              // stride hi, dim1_stride lo
  g1[7] = 0;
  v4i g2 = {0, 0, 0, 0}, g3 = {0, 0, 0, 0};
#if defined(__clang_major__) && __clang_major__ >= 23
  v8i g4 = {0, 0, 0, 0, 0, 0, 0, 0};
  __builtin_amdgcn_tensor_load_to_lds(g0, g1, g2, g3, g4, 0);
#else
  __builtin_amdgcn_tensor_load_to_lds(g0, g1, g2, g3, 0);
#endif
}
#else
// fallback: per-wave async copy of one 128x32 bf16 tile
__device__ __forceinline__ void stage_tile_async(const bf16* __restrict__ g, size_t ld,
                                                 int row0, int k0, bf16* lds, int tid) {
#pragma unroll
  for (int c = tid; c < 512; c += 256) {
    int r = c >> 2, cc = (c & 3) << 3;
    async_b128((unsigned)(size_t)(lds + r * LDP + cc),
               g + (size_t)(row0 + r) * ld + k0 + cc);
  }
}
#endif

// ---------------- TDM-staged GEMM ----------------
// C[M,N](f32) = A[M,K](bf16 row-major) @ Wt[N,K](bf16 pre-transposed)
// mode 0: C = acc + bias;  mode 1: Cb = bf16(gelu_tanh(C)*acc);  mode 2: Cb = bf16(acc)
__global__ __launch_bounds__(256) void k_gemm(const bf16* __restrict__ A,
                                              const bf16* __restrict__ Wt,
                                              float* __restrict__ C,
                                              bf16* __restrict__ Cb,
                                              const float* __restrict__ bias,
                                              int M, int N, int K, int mode) {
  __shared__ __align__(16) bf16 sA[2][BM][LDP];  // 2 x 10 KB
  __shared__ __align__(16) bf16 sB[2][BN][LDP];  // 2 x 10 KB
  const int tid  = threadIdx.x;
  const int lane = tid & 31;
  const int wave = tid >> 5;
  const int wm = wave & 3, wn = wave >> 2;       // 4x2 wave grid -> 32x64 per wave
  const int bm0 = blockIdx.y * BM, bn0 = blockIdx.x * BN;

  floatx8 acc[2][4];
#pragma unroll
  for (int i = 0; i < 2; i++)
#pragma unroll
    for (int j = 0; j < 4; j++) acc[i][j] = zero8();

#ifdef HAS_TDM
  if (wave == 0) {   // wave 0 drives the Tensor Data Mover
    tdm_load_tile(A  + (size_t)bm0 * K, (unsigned)(size_t)&sA[0][0][0], K);
    tdm_load_tile(Wt + (size_t)bn0 * K, (unsigned)(size_t)&sB[0][0][0], K);
    __builtin_amdgcn_s_wait_tensorcnt(0);
  }
#else
  stage_tile_async(A,  K, bm0, 0, &sA[0][0][0], tid);
  stage_tile_async(Wt, K, bn0, 0, &sB[0][0][0], tid);
  wait_async0();
#endif
  __syncthreads();

  int buf = 0;
  for (int k0 = 0; k0 < K; k0 += BK) {
    const bool more = (k0 + BK < K);
    if (more) {  // overlap next tile's DMA with this tile's WMMAs
#ifdef HAS_TDM
      if (wave == 0) {
        tdm_load_tile(A  + (size_t)bm0 * K + k0 + BK, (unsigned)(size_t)&sA[buf ^ 1][0][0], K);
        tdm_load_tile(Wt + (size_t)bn0 * K + k0 + BK, (unsigned)(size_t)&sB[buf ^ 1][0][0], K);
      }
#else
      stage_tile_async(A,  K, bm0, k0 + BK, &sA[buf ^ 1][0][0], tid);
      stage_tile_async(Wt, K, bn0, k0 + BK, &sB[buf ^ 1][0][0], tid);
#endif
    }
    bf16x16 aF0 = frag_a(&sA[buf][0][0], LDP, wm * 32,      0, lane);
    bf16x16 aF1 = frag_a(&sA[buf][0][0], LDP, wm * 32 + 16, 0, lane);
#pragma unroll
    for (int ni = 0; ni < 4; ni++) {
      bf16x16 bF = frag_b(&sB[buf][0][0], LDP, wn * 64 + ni * 16, 0, lane);
      acc[0][ni] = wmma_bf16(aF0, bF, acc[0][ni]);
      acc[1][ni] = wmma_bf16(aF1, bF, acc[1][ni]);
    }
    if (more) {
#ifdef HAS_TDM
      if (wave == 0) __builtin_amdgcn_s_wait_tensorcnt(0);
#else
      wait_async0();
#endif
    }
    __syncthreads();
    buf ^= 1;
  }

#pragma unroll
  for (int mi = 0; mi < 2; mi++) {
    int mb = bm0 + wm * 32 + mi * 16 + ((lane >> 4) << 3);
#pragma unroll
    for (int ni = 0; ni < 4; ni++) {
      int n = bn0 + wn * 64 + ni * 16 + (lane & 15);
      float bi = (bias != nullptr) ? bias[n] : 0.f;
#pragma unroll
      for (int r = 0; r < 8; r++) {
        size_t idx = (size_t)(mb + r) * N + n;
        float a = acc[mi][ni][r];
        if (mode == 1)      Cb[idx] = f2bf(gelu_tanh(C[idx]) * a);
        else if (mode == 2) Cb[idx] = f2bf(a);
        else                C[idx]  = a + bi;
      }
    }
  }
}

// ---------------- Flash attention with softcap (WMMA, async-staged bf16) ----------------
__global__ __launch_bounds__(128) void k_attn(const bf16* __restrict__ Q,
                                              const bf16* __restrict__ K,
                                              const bf16* __restrict__ Vv,
                                              bf16* __restrict__ O,
                                              int sliding) {
  __shared__ __align__(16) bf16 sQ[64][64];   // [q][d], async-filled
  __shared__ __align__(16) bf16 sK[32][64];   // [kv][d] == B-layout [n=kv][k=d]
  __shared__ bf16 sVt[64][34];                // [d][kv] == B-layout [n=d][k=kv]
  __shared__ bf16 sP[4][16][34];              // per-wave P tile, A-layout

  const int tid = threadIdx.x, lane = tid & 31, wave = tid >> 5;
  const int q0 = blockIdx.x * 64, h = blockIdx.y, b = blockIdx.z;
  const size_t rowbase = (size_t)(b * S_) * E_ + h * D_;

  // async-stage the Q tile (64x64 bf16 = 512 16B chunks)
  for (int c = tid; c < 512; c += 128) {
    int r = c >> 3, cc = (c & 7) << 3;
    async_b128((unsigned)(size_t)&sQ[r][cc], Q + rowbase + (size_t)(q0 + r) * E_ + cc);
  }
  wait_async0();
  __syncthreads();

  float mrow[8], lrow[8];
  floatx8 oacc[4];
#pragma unroll
  for (int r = 0; r < 8; r++) { mrow[r] = NEGINF; lrow[r] = 0.f; }
#pragma unroll
  for (int dt = 0; dt < 4; dt++) oacc[dt] = zero8();

  int kv_lo = 0;
  if (sliding) { int t = q0 - (SW_ - 1); if (t < 0) t = 0; kv_lo = t & ~31; }

  const int rbase = (lane >> 4) << 3;
  const int ncol  = lane & 15;

  for (int kv0 = kv_lo; kv0 < q0 + 64; kv0 += 32) {
    if (kv0 + 32 < q0 + 64) {  // WGP-scope prefetch of next K/V block
      __builtin_prefetch(K  + rowbase + (size_t)(kv0 + 32 + (tid & 31)) * E_, 0, 3);
      __builtin_prefetch(Vv + rowbase + (size_t)(kv0 + 32 + (tid & 31)) * E_, 0, 3);
    }
    __syncthreads();
    // async K tile (32x64 = 256 chunks) overlapped with manual V transpose
    for (int c = tid; c < 256; c += 128) {
      int r = c >> 3, cc = (c & 7) << 3;
      async_b128((unsigned)(size_t)&sK[r][cc], K + rowbase + (size_t)(kv0 + r) * E_ + cc);
    }
    for (int c = tid; c < 256; c += 128) {
      int r = c >> 3, cc = (c & 7) << 3;
      uint4 w = *(const uint4*)(Vv + rowbase + (size_t)(kv0 + r) * E_ + cc);
      const unsigned short* u = (const unsigned short*)&w;
#pragma unroll
      for (int j = 0; j < 8; j++) *(unsigned short*)&sVt[cc + j][r] = u[j];
    }
    wait_async0();
    __syncthreads();

    bf16x16 aq0 = frag_a(&sQ[0][0], 64, wave * 16,  0, lane);
    bf16x16 aq1 = frag_a(&sQ[0][0], 64, wave * 16, 32, lane);
    floatx8 s0 = zero8(), s1 = zero8();
    s0 = wmma_bf16(aq0, frag_b(&sK[0][0], 64,  0,  0, lane), s0);
    s0 = wmma_bf16(aq1, frag_b(&sK[0][0], 64,  0, 32, lane), s0);
    s1 = wmma_bf16(aq0, frag_b(&sK[0][0], 64, 16,  0, lane), s1);
    s1 = wmma_bf16(aq1, frag_b(&sK[0][0], 64, 16, 32, lane), s1);

#pragma unroll
    for (int r = 0; r < 8; r++) {
      int q = q0 + wave * 16 + rbase + r;
      float v0 = s0[r] * 0.125f; v0 = 50.f * tanhf(v0 * 0.02f);
      float v1 = s1[r] * 0.125f; v1 = 50.f * tanhf(v1 * 0.02f);
      int k0i = kv0 + ncol, k1i = kv0 + 16 + ncol;
      bool ok0 = (k0i <= q) && (!sliding || (q - k0i) < SW_);
      bool ok1 = (k1i <= q) && (!sliding || (q - k1i) < SW_);
      v0 = ok0 ? v0 : NEGINF;
      v1 = ok1 ? v1 : NEGINF;
      float rm = fmaxf(v0, v1);
      rm = fmaxf(rm, __shfl_xor(rm, 1, 32));
      rm = fmaxf(rm, __shfl_xor(rm, 2, 32));
      rm = fmaxf(rm, __shfl_xor(rm, 4, 32));
      rm = fmaxf(rm, __shfl_xor(rm, 8, 32));
      float mnew = fmaxf(mrow[r], rm);
      float corr = __expf(mrow[r] - mnew);
      float p0 = ok0 ? __expf(v0 - mnew) : 0.f;
      float p1 = ok1 ? __expf(v1 - mnew) : 0.f;
      float rs = p0 + p1;
      rs += __shfl_xor(rs, 1, 32); rs += __shfl_xor(rs, 2, 32);
      rs += __shfl_xor(rs, 4, 32); rs += __shfl_xor(rs, 8, 32);
      lrow[r] = lrow[r] * corr + rs;
      mrow[r] = mnew;
#pragma unroll
      for (int dt = 0; dt < 4; dt++) oacc[dt][r] *= corr;
      sP[wave][rbase + r][ncol]      = f2bf(p0);
      sP[wave][rbase + r][16 + ncol] = f2bf(p1);
    }

    bf16x16 ap = frag_a(&sP[wave][0][0], 34, 0, 0, lane);
#pragma unroll
    for (int dt = 0; dt < 4; dt++) {
      bf16x16 bv = frag_b(&sVt[0][0], 34, dt * 16, 0, lane);
      oacc[dt] = wmma_bf16(ap, bv, oacc[dt]);
    }
  }

#pragma unroll
  for (int r = 0; r < 8; r++) {
    float inv = 1.f / lrow[r];
    int q = q0 + wave * 16 + rbase + r;
#pragma unroll
    for (int dt = 0; dt < 4; dt++)
      O[rowbase + (size_t)q * E_ + dt * 16 + ncol] = f2bf(oacc[dt][r] * inv);
  }
}

// ---------------- glue kernels ----------------
__global__ __launch_bounds__(256) void k_embed(const int* __restrict__ ids,
                                               const float* __restrict__ emb,
                                               float* __restrict__ x) {
  int row = blockIdx.x, tid = threadIdx.x;
  const float* e = emb + (size_t)ids[row] * E_;
  float* d = x + (size_t)row * E_;
  d[tid] = e[tid]; d[tid + 256] = e[tid + 256]; d[tid + 512] = e[tid + 512];
}

// Gemma RMSNorm (scale = 1+w).  dstb!=null: dstb = bf16(rms(src));
// else: dstf = rms(src) (add=0) or dstf += rms(src) (add=1)
__global__ __launch_bounds__(256) void k_rmsnorm(float* __restrict__ dstf,
                                                 bf16* __restrict__ dstb,
                                                 const float* __restrict__ src,
                                                 const float* __restrict__ w, int add) {
  __shared__ float red[256];
  int row = blockIdx.x, tid = threadIdx.x;
  const float* x = src + (size_t)row * E_;
  float v0 = x[tid], v1 = x[tid + 256], v2 = x[tid + 512];
  red[tid] = v0 * v0 + v1 * v1 + v2 * v2;
  __syncthreads();
  for (int off = 128; off > 0; off >>= 1) {
    if (tid < off) red[tid] += red[tid + off];
    __syncthreads();
  }
  float inv = rsqrtf(red[0] / (float)E_ + 1e-6f);
  float o0 = v0 * inv * (1.f + w[tid]);
  float o1 = v1 * inv * (1.f + w[tid + 256]);
  float o2 = v2 * inv * (1.f + w[tid + 512]);
  if (dstb) {
    bf16* d = dstb + (size_t)row * E_;
    d[tid] = f2bf(o0); d[tid + 256] = f2bf(o1); d[tid + 512] = f2bf(o2);
  } else {
    float* d = dstf + (size_t)row * E_;
    if (add) { d[tid] += o0; d[tid + 256] += o1; d[tid + 512] += o2; }
    else     { d[tid]  = o0; d[tid + 256]  = o1; d[tid + 512]  = o2; }
  }
}

__global__ void k_ropetab(float* __restrict__ cosb, float* __restrict__ sinb,
                          const float* __restrict__ pos) {
  int row = blockIdx.x, j = threadIdx.x;  // 32 threads = D/2
  float pv = pos ? pos[row] : (float)(row & (S_ - 1));
  float inv = __powf(10000.f, -(float)(2 * j) / (float)D_);
  float f = pv * inv;
  float c = __cosf(f), s = __sinf(f);
  size_t o = (size_t)row * D_;
  cosb[o + j] = c; cosb[o + j + 32] = c;
  sinb[o + j] = s; sinb[o + j + 32] = s;
}

// RoPE, f32 in -> bf16 out (feeds async-staged attention directly)
__global__ __launch_bounds__(384) void k_rope(const float* __restrict__ x,
                                              bf16* __restrict__ o,
                                              const float* __restrict__ cosb,
                                              const float* __restrict__ sinb) {
  int row = blockIdx.x, t = threadIdx.x;
  int hh = t >> 5, j = t & 31;
  size_t base = (size_t)row * E_ + hh * D_;
  float x1 = x[base + j], x2 = x[base + j + 32];
  float c = cosb[(size_t)row * D_ + j], s = sinb[(size_t)row * D_ + j];
  o[base + j]      = f2bf(x1 * c - x2 * s);
  o[base + j + 32] = f2bf(x2 * c + x1 * s);
}

// transpose-convert: Wt[n*K+k] = bf16(W[k*N+n]);  K,N multiples of 32
__global__ void k_cvt_t(const float* __restrict__ W, bf16* __restrict__ Wt, int K, int N) {
  __shared__ float t[32][33];
  int k0 = blockIdx.x * 32, n0 = blockIdx.y * 32;
  int tx = threadIdx.x, ty = threadIdx.y;  // 32 x 8
  for (int i = ty; i < 32; i += 8) t[i][tx] = W[(size_t)(k0 + i) * N + n0 + tx];
  __syncthreads();
  for (int i = ty; i < 32; i += 8) Wt[(size_t)(n0 + i) * K + k0 + tx] = f2bf(t[tx][i]);
}

__global__ __launch_bounds__(256) void k_cvtbf(const float* __restrict__ s,
                                               bf16* __restrict__ d) {
  int row = blockIdx.x, tid = threadIdx.x;
  size_t o = (size_t)row * E_;
  d[o + tid]       = f2bf(s[o + tid]);
  d[o + tid + 256] = f2bf(s[o + tid + 256]);
  d[o + tid + 512] = f2bf(s[o + tid + 512]);
}

__global__ void k_scan(const int* __restrict__ pres, int* __restrict__ dst,
                       int* __restrict__ updst, int* __restrict__ gstart,
                       int* __restrict__ gcnt) {
  int b = blockIdx.x;
  for (int g = 0; g < S_; g++) { gstart[b * S_ + g] = 0; gcnt[b * S_ + g] = 0; }
  int c = 0, uc = 0;
  for (int s = 0; s < S_; s++) {
    int gg = (s == 0) ? 1 : (pres[b * S_ + s] != 0);
    if (gg) { gstart[b * S_ + c] = s; c++; }
    dst[b * S_ + s] = c - 1;
    gcnt[b * S_ + (c - 1)]++;
    int ug = (s < S_ - 1) ? (pres[b * S_ + s + 1] != 0) : 1;
    uc += ug;
    updst[b * S_ + s] = uc - 1;
  }
}

__global__ __launch_bounds__(256) void k_downsample(const float* __restrict__ x,
                                                    const int* __restrict__ gstart,
                                                    const int* __restrict__ gcnt,
                                                    float* __restrict__ xds,
                                                    float* __restrict__ xm,
                                                    float* __restrict__ posds) {
  int b = blockIdx.y, g = blockIdx.x, tid = threadIdx.x;
  int row = b * S_ + g;
  int st = gstart[row], n = gcnt[row];
  float invn = (n > 0) ? 1.f / (float)n : 0.f;
  for (int e = tid; e < E_; e += 256) {
    float s = 0.f;
    for (int i = 0; i < n; i++) s += x[(size_t)(b * S_ + st + i) * E_ + e];
    float v = s * invn;
    xds[(size_t)row * E_ + e] = v;
    xm[(size_t)row * E_ + e]  = v;
  }
  if (tid == 0) posds[row] = (n > 0) ? ((float)st + 0.5f * (float)(n - 1)) : 0.f;
}

__global__ __launch_bounds__(256) void k_upsample(float* __restrict__ x,
                                                  const float* __restrict__ ym,
                                                  const float* __restrict__ xds,
                                                  const int* __restrict__ updst) {
  int row = blockIdx.x, tid = threadIdx.x;
  int b = row >> 10;  // S_ == 1024
  int dr = b * S_ + updst[row];
  size_t ro = (size_t)row * E_, so = (size_t)dr * E_;
  x[ro + tid]       += ym[so + tid]       - xds[so + tid];
  x[ro + tid + 256] += ym[so + tid + 256] - xds[so + tid + 256];
  x[ro + tid + 512] += ym[so + tid + 512] - xds[so + tid + 512];
}

__global__ __launch_bounds__(256) void k_logsoftmax(float* __restrict__ y) {
  __shared__ float red[256];
  int row = blockIdx.x, tid = threadIdx.x;  // V_ == 256
  float* p = y + (size_t)row * V_;
  float v = p[tid];
  red[tid] = v; __syncthreads();
  for (int off = 128; off > 0; off >>= 1) {
    if (tid < off) red[tid] = fmaxf(red[tid], red[tid + off]);
    __syncthreads();
  }
  float mx = red[0]; __syncthreads();
  float e = __expf(v - mx);
  red[tid] = e; __syncthreads();
  for (int off = 128; off > 0; off >>= 1) {
    if (tid < off) red[tid] += red[tid + off];
    __syncthreads();
  }
  float ls = __logf(red[0]);
  p[tid] = v - mx - ls;
}

// ---------------- host-side orchestration ----------------
struct StackPtrs { const float *wq,*wk,*wv,*wo,*wg,*wu,*wd,*l1,*l2,*l3,*l4; };
struct WtStack   { bf16 *wq,*wk,*wv,*wo,*wg,*wu,*wd; };

static void build_wt(const StackPtrs& p, const WtStack& w, int n, int inter, hipStream_t st) {
  const size_t ee = (size_t)E_ * E_, ei = (size_t)E_ * inter;
  dim3 blk(32, 8);
  for (int li = 0; li < n; li++) {
    k_cvt_t<<<dim3(E_ / 32, E_ / 32), blk, 0, st>>>(p.wq + li * ee, w.wq + li * ee, E_, E_);
    k_cvt_t<<<dim3(E_ / 32, E_ / 32), blk, 0, st>>>(p.wk + li * ee, w.wk + li * ee, E_, E_);
    k_cvt_t<<<dim3(E_ / 32, E_ / 32), blk, 0, st>>>(p.wv + li * ee, w.wv + li * ee, E_, E_);
    k_cvt_t<<<dim3(E_ / 32, E_ / 32), blk, 0, st>>>(p.wo + li * ee, w.wo + li * ee, E_, E_);
    k_cvt_t<<<dim3(E_ / 32, inter / 32), blk, 0, st>>>(p.wg + li * ei, w.wg + li * ei, E_, inter);
    k_cvt_t<<<dim3(E_ / 32, inter / 32), blk, 0, st>>>(p.wu + li * ei, w.wu + li * ei, E_, inter);
    k_cvt_t<<<dim3(inter / 32, E_ / 32), blk, 0, st>>>(p.wd + li * ei, w.wd + li * ei, inter, E_);
  }
}

static void run_layer(const StackPtrs& p, const WtStack& w, int li, int inter, int sliding,
                      float* x, float* qf, float* kf, float* ff,
                      bf16* hb, bf16* ob, bf16* ffb, bf16* qb, bf16* kb, bf16* vb,
                      const float* cosb, const float* sinb, hipStream_t st) {
  const size_t ee = (size_t)E_ * E_, ei = (size_t)E_ * inter;
  const bf16 *wq = w.wq + li * ee, *wk = w.wk + li * ee, *wv = w.wv + li * ee,
             *wo = w.wo + li * ee, *wg = w.wg + li * ei, *wu = w.wu + li * ei,
             *wd = w.wd + li * ei;
  const float *l1 = p.l1 + (size_t)li * E_, *l2 = p.l2 + (size_t)li * E_,
              *l3 = p.l3 + (size_t)li * E_, *l4 = p.l4 + (size_t)li * E_;

  dim3 gE(E_ / BN, BS_ / BM), gI(inter / BN, BS_ / BM);

  k_rmsnorm<<<BS_, 256, 0, st>>>(nullptr, hb, x, l1, 0);
  k_gemm<<<gE, 256, 0, st>>>(hb, wq, qf, nullptr, nullptr, BS_, E_, E_, 0);
  k_gemm<<<gE, 256, 0, st>>>(hb, wk, kf, nullptr, nullptr, BS_, E_, E_, 0);
  k_gemm<<<gE, 256, 0, st>>>(hb, wv, nullptr, vb, nullptr, BS_, E_, E_, 2);  // V -> bf16
  k_rope<<<BS_, 384, 0, st>>>(qf, qb, cosb, sinb);
  k_rope<<<BS_, 384, 0, st>>>(kf, kb, cosb, sinb);
  k_attn<<<dim3(S_ / 64, H_, B_), 128, 0, st>>>(qb, kb, vb, ob, sliding);
  k_gemm<<<gE, 256, 0, st>>>(ob, wo, qf, nullptr, nullptr, BS_, E_, E_, 0);
  k_rmsnorm<<<BS_, 256, 0, st>>>(x, nullptr, qf, l2, 1);
  k_rmsnorm<<<BS_, 256, 0, st>>>(nullptr, hb, x, l3, 0);
  k_gemm<<<gI, 256, 0, st>>>(hb, wg, ff, nullptr, nullptr, BS_, inter, E_, 0);
  k_gemm<<<gI, 256, 0, st>>>(hb, wu, ff, ffb, nullptr, BS_, inter, E_, 1);  // ffb=gelu(ff)*u
  k_gemm<<<gE, 256, 0, st>>>(ffb, wd, qf, nullptr, nullptr, BS_, E_, inter, 0);
  k_rmsnorm<<<BS_, 256, 0, st>>>(x, nullptr, qf, l4, 1);
}

extern "C" void kernel_launch(void* const* d_in, const int* in_sizes, int n_in,
                              void* d_out, int out_size, void* d_ws, size_t ws_size,
                              hipStream_t stream) {
  (void)in_sizes; (void)n_in; (void)out_size; (void)ws_size;

  const int*   ids   = (const int*)d_in[0];
  const int*   pres  = (const int*)d_in[1];
  const float* emb   = (const float*)d_in[2];
  // d_in[3], d_in[4]: gate_w/gate_b -- gate outputs are discarded by the reference.
  const float* out_w = (const float*)d_in[5];
  const float* out_b = (const float*)d_in[6];

  auto mk = [&](int base) {
    StackPtrs s;
    s.wq = (const float*)d_in[base + 0];  s.wk = (const float*)d_in[base + 1];
    s.wv = (const float*)d_in[base + 2];  s.wo = (const float*)d_in[base + 3];
    s.wg = (const float*)d_in[base + 4];  s.wu = (const float*)d_in[base + 5];
    s.wd = (const float*)d_in[base + 6];  s.l1 = (const float*)d_in[base + 7];
    s.l2 = (const float*)d_in[base + 8];  s.l3 = (const float*)d_in[base + 9];
    s.l4 = (const float*)d_in[base + 10];
    return s;
  };
  StackPtrs down = mk(7), mid = mk(18), up = mk(29);

  char* wsp = (char*)d_ws;
  auto take = [&](size_t bytes) -> void* {
    void* p = (void*)wsp;
    wsp += (bytes + 255) & ~(size_t)255;
    return p;
  };
  const size_t fBSE = sizeof(float) * (size_t)BS_ * E_;
  const size_t bBSE = sizeof(bf16) * (size_t)BS_ * E_;
  float* x    = (float*)take(fBSE);
  float* xm   = (float*)take(fBSE);
  float* xds  = (float*)take(fBSE);
  float* qf   = (float*)take(fBSE);
  float* kf   = (float*)take(fBSE);
  float* ff   = (float*)take(sizeof(float) * (size_t)BS_ * IMID_);
  bf16*  hb   = (bf16*)take(bBSE);
  bf16*  ob   = (bf16*)take(bBSE);
  bf16*  xb   = (bf16*)take(bBSE);
  bf16*  qb   = (bf16*)take(bBSE);
  bf16*  kb   = (bf16*)take(bBSE);
  bf16*  vb   = (bf16*)take(bBSE);
  bf16*  ffb  = (bf16*)take(sizeof(bf16) * (size_t)BS_ * IMID_);
  float* cosb = (float*)take(sizeof(float) * (size_t)BS_ * D_);
  float* sinb = (float*)take(sizeof(float) * (size_t)BS_ * D_);
  float* cos2 = (float*)take(sizeof(float) * (size_t)BS_ * D_);
  float* sin2 = (float*)take(sizeof(float) * (size_t)BS_ * D_);
  float* posds  = (float*)take(sizeof(float) * BS_);
  int*   dst    = (int*)take(sizeof(int) * BS_);
  int*   updst  = (int*)take(sizeof(int) * BS_);
  int*   gstart = (int*)take(sizeof(int) * BS_);
  int*   gcnt   = (int*)take(sizeof(int) * BS_);

  auto mkwt = [&](int n, int inter) {
    WtStack w;
    const size_t ee = (size_t)E_ * E_, ei = (size_t)E_ * inter;
    w.wq = (bf16*)take(sizeof(bf16) * n * ee);
    w.wk = (bf16*)take(sizeof(bf16) * n * ee);
    w.wv = (bf16*)take(sizeof(bf16) * n * ee);
    w.wo = (bf16*)take(sizeof(bf16) * n * ee);
    w.wg = (bf16*)take(sizeof(bf16) * n * ei);
    w.wu = (bf16*)take(sizeof(bf16) * n * ei);
    w.wd = (bf16*)take(sizeof(bf16) * n * ei);
    return w;
  };
  WtStack wdn = mkwt(2, E_), wmd = mkwt(6, IMID_), wup = mkwt(2, E_);
  bf16* wout = (bf16*)take(sizeof(bf16) * (size_t)E_ * V_);

  // once-per-pass weight transpose+convert to bf16 [N,K]
  build_wt(down, wdn, 2, E_, stream);
  build_wt(mid,  wmd, 6, IMID_, stream);
  build_wt(up,   wup, 2, E_, stream);
  k_cvt_t<<<dim3(E_ / 32, V_ / 32), dim3(32, 8), 0, stream>>>(out_w, wout, E_, V_);

  k_embed<<<BS_, 256, 0, stream>>>(ids, emb, x);
  k_ropetab<<<BS_, 32, 0, stream>>>(cosb, sinb, nullptr);

  for (int l = 0; l < 2; l++)
    run_layer(down, wdn, l, E_, 1, x, qf, kf, ff, hb, ob, ffb, qb, kb, vb, cosb, sinb, stream);

  k_scan<<<B_, 1, 0, stream>>>(pres, dst, updst, gstart, gcnt);
  k_downsample<<<dim3(S_, B_), 256, 0, stream>>>(x, gstart, gcnt, xds, xm, posds);
  k_ropetab<<<BS_, 32, 0, stream>>>(cos2, sin2, posds);

  for (int l = 0; l < 6; l++)
    run_layer(mid, wmd, l, IMID_, 0, xm, qf, kf, ff, hb, ob, ffb, qb, kb, vb, cos2, sin2, stream);

  k_upsample<<<BS_, 256, 0, stream>>>(x, xm, xds, updst);

  for (int l = 0; l < 2; l++)
    run_layer(up, wup, l, E_, 1, x, qf, kf, ff, hb, ob, ffb, qb, kb, vb, cosb, sinb, stream);

  k_cvtbf<<<BS_, 256, 0, stream>>>(x, xb);
  k_gemm<<<dim3(V_ / BN, BS_ / BM), 256, 0, stream>>>(xb, wout, (float*)d_out, nullptr,
                                                      out_b, BS_, V_, E_, 0);
  k_logsoftmax<<<BS_, 256, 0, stream>>>((float*)d_out);
}
